// TreeEncoder_34626026341055
// MI455X (gfx1250) — matrix operand order repaired and checked by reading, not compiled, not run on hardware
//
#include <hip/hip_runtime.h>
#include <hip/hip_bf16.h>

// Fused TreeEncoder (no-children path): emb GEMM + 2 layers of 3 gate GEMMs +
// activations + batch-mean. bf16 WMMA (f32 accumulate). A prep kernel packs
// all weights (fp32->bf16, transposed [n][k]) plus pre-summed f32 biases into
// d_ws; the main kernel DMA-stages the whole blob into LDS with ONE
// tensor_load_to_lds (TDM) and runs entirely out of LDS/registers.
// Nonlinearities use native v_exp_f32/v_rcp_f32; bf16 packing uses
// v_cvt_pk_bf16_f32.

typedef __bf16 bf16_t;
typedef __attribute__((ext_vector_type(16))) __bf16 v16bf;
typedef __attribute__((ext_vector_type(8)))  float  v8f;
typedef __attribute__((ext_vector_type(4)))  unsigned int v4u;
typedef __attribute__((ext_vector_type(8)))  int v8i;
typedef __attribute__((ext_vector_type(4)))  int v4i;

#define B_      256
#define N_      512
#define F_      64
#define H_      128
#define L_      2
#define THREADS 512
#define WAVES   16           // 512 threads / wave32
#define ITERS   2            // 512 rows / (16 waves * 16 rows)
#define WELEMS  (H_ * F_ + L_ * 3 * H_ * H_)   // 106496 bf16 = 208 KB
#define NBIAS   ((1 + L_ * 3) * H_)            // 896 f32: emb + 6 combined
#define BLOB_BYTES (WELEMS * 2 + NBIAS * 4)    // 216576 B, 8B-aligned

#define LOG2E 1.4426950408889634f

__device__ __forceinline__ float sigm(float x) {
  return __builtin_amdgcn_rcpf(1.0f + __builtin_amdgcn_exp2f(-x * LOG2E));
}
__device__ __forceinline__ float tanh_(float x) {
  return 1.0f - 2.0f * __builtin_amdgcn_rcpf(1.0f + __builtin_amdgcn_exp2f(x * (2.0f * LOG2E)));
}

// hardware packed f32 -> 2x bf16 convert (RNE)
__device__ __forceinline__ unsigned cvt_pk_bf16(float a, float b) {
  unsigned r;
  asm("v_cvt_pk_bf16_f32 %0, %1, %2" : "=v"(r) : "v"(a), "v"(b));
  return r;
}
__device__ __forceinline__ void store_bf16(bf16_t* p, float v) {
  unsigned pk = cvt_pk_bf16(v, v);
  *(unsigned short*)p = (unsigned short)pk;   // ds_store_b16 uses low half
}

__device__ __forceinline__ bf16_t f2bf(float f) {   // prep-kernel only
  unsigned u = __builtin_bit_cast(unsigned, f);
  u += 0x7fffu + ((u >> 16) & 1u);
  unsigned short h = (unsigned short)(u >> 16);
  return __builtin_bit_cast(bf16_t, h);
}

// Activations are stored unpermuted so a 16x32 bf16 A fragment is ONE
// contiguous 32B LDS read per lane: lanes<16 get K 0..15, lanes>=16
// get K 16..31 (matching the B fragment K order).
__device__ __forceinline__ int permK(int k) {
  return k;
}

__device__ __forceinline__ v16bf load_fragA(const bf16_t* rowp, int ks, int lane) {
  return *(const v16bf*)(rowp + ks * 32 + ((lane >= 16) ? 16 : 0));
}

// B fragment (32x16 bf16, N=lane&15): lanes<16 hold K 0..15 (+32*ks),
// lanes>=16 hold K 16..31; weights packed transposed [n][k] -> one contiguous
// 32B LDS read per lane, no permutation needed.
__device__ __forceinline__ v16bf load_fragB(const bf16_t* wt, int K, int nt, int ks, int lane) {
  int n  = nt * 16 + (lane & 15);
  int kb = ks * 32 + ((lane >= 16) ? 16 : 0);
  return *(const v16bf*)(wt + n * K + kb);
}

// ---- prep: pack weights (bf16, transposed) + combined biases (f32) --------
// Blob: [0, 8192)                 emb_W^T  [n=128][k=64]      (bf16)
//       [8192 + m*16384)          gate m=(l*3+g)^T [128][128] (bf16)
//       then f32: [0,128) embB; [128 + m*128 + c) bW[m][c]+bU[m][c]
__global__ void prep_weights(const float* __restrict__ embW,
                             const float* __restrict__ embB,
                             const float* __restrict__ Wi,
                             const float* __restrict__ Wo,
                             const float* __restrict__ Wc,
                             const float* __restrict__ bWi, const float* __restrict__ bUi,
                             const float* __restrict__ bWo, const float* __restrict__ bUo,
                             const float* __restrict__ bWc, const float* __restrict__ bUc,
                             bf16_t* __restrict__ ws) {
  int e = blockIdx.x * blockDim.x + threadIdx.x;
  if (e < H_ * F_) {
    int n = e >> 6, k = e & 63;
    ws[e] = f2bf(embW[k * H_ + n]);
  } else if (e < WELEMS) {
    int e2 = e - H_ * F_;
    int m = e2 >> 14;                 // matrix 0..5
    int r = e2 & 16383;
    int n = r >> 7, k = r & 127;
    int l = m / 3, g = m % 3;
    const float* W = (g == 0) ? Wi : (g == 1) ? Wo : Wc;
    ws[e] = f2bf(W[l * H_ * H_ + k * H_ + n]);
  } else if (e < WELEMS + NBIAS) {
    float* bias = (float*)(ws + WELEMS);
    int j = e - WELEMS;
    if (j < H_) {
      bias[j] = embB[j];
    } else {
      int m = (j - H_) >> 7;          // 0..5
      int c = (j - H_) & 127;
      int l = m / 3, g = m % 3;
      const float* bW = (g == 0) ? bWi : (g == 1) ? bWo : bWc;
      const float* bU = (g == 0) ? bUi : (g == 1) ? bUo : bUc;
      bias[j] = bW[l * H_ + c] + bU[l * H_ + c];
    }
  }
}

// ---- main fused kernel: 1 workgroup = 1 batch -----------------------------
__global__ __launch_bounds__(THREADS)
void tree_encoder_fused(const float* __restrict__ feat,   // [B,N,F]
                        const bf16_t* __restrict__ wsrc,  // packed blob
                        float* __restrict__ out)          // [B,H]
{
  __shared__ bf16_t wlds[WELEMS + NBIAS * 2]; // 211.5 KB: weights + f32 biases
  __shared__ bf16_t hbuf[WAVES * 16 * H_];    //  64 KB: per-wave act tiles
  __shared__ float  pacc[WAVES * 2 * H_];     //  16 KB: deterministic reduce

  const int tid    = threadIdx.x;
  const int lane   = tid & 31;
  const int wave   = tid >> 5;
  const int b      = blockIdx.x;
  const int c_lane = lane & 15;
  const int rbase  = (lane >= 16) ? 8 : 0;

  // ---- TDM: one DMA of the whole blob (weights + biases) global -> LDS ----
  if (wave == 0) {
    unsigned long long ga = (unsigned long long)(size_t)wsrc;
    v4u g0;
    g0[0] = 1u;                                        // count=1, user mode
    g0[1] = (unsigned)(size_t)(void*)&wlds[0];         // lds_addr (bytes)
    g0[2] = (unsigned)ga;                              // global_addr lo
    g0[3] = (unsigned)((ga >> 32) & 0x1FFFFFFu) | 0x80000000u;  // hi + type=2
    const unsigned nq = (unsigned)(BLOB_BYTES / 8);    // 8-byte elems: 27072
    v8i g1 = {0, 0, 0, 0, 0, 0, 0, 0};
    g1[0] = (int)(3u << 16);                           // data_size = 8B
    g1[1] = (int)((nq & 0xffffu) << 16);               // tensor_dim0 lo16
    g1[2] = (int)(((nq >> 16) & 0xffffu) | (1u << 16)); // dim0 hi | tensor_dim1=1
    g1[3] = (int)((nq & 0xffffu) << 16);               // tile_dim0
    g1[4] = 1;                                         // tile_dim1 = 1
    g1[5] = (int)nq;                                   // tensor_dim0_stride
    v4i gz4 = {0, 0, 0, 0};
    v8i gz8 = {0, 0, 0, 0, 0, 0, 0, 0};
    __builtin_amdgcn_tensor_load_to_lds(g0, g1, gz4, gz4, gz8, 0);
    __builtin_amdgcn_s_wait_tensorcnt((short)0);
  }
  __syncthreads();

  const bf16_t* embT  = wlds;                          // [128][64]
  const float*  biasL = (const float*)(wlds + WELEMS); // emb + 6 combined
  bf16_t* myh = hbuf + wave * (16 * H_);
  float wsum[8];
  #pragma unroll
  for (int nt = 0; nt < 8; ++nt) wsum[nt] = 0.0f;

  for (int it = 0; it < ITERS; ++it) {
    const int row0 = it * (WAVES * 16) + wave * 16;
    const float* fp = feat + ((size_t)b * N_ + row0) * F_;

    // ---- features -> per-wave LDS tile [16][F_] (row stride H_, K unpermuted)
    #pragma unroll
    for (int t = 0; t < 8; ++t) {
      int e = lane + 32 * t;                 // float4 index, 16 rows x 16 quads
      int r = e >> 4;
      int q = e & 15;
      float4 v = *(const float4*)(fp + e * 4);   // coalesced b128 loads
      uint2 pk;
      pk.x = cvt_pk_bf16(v.x, v.y);
      pk.y = cvt_pk_bf16(v.z, v.w);
      *(uint2*)(myh + r * H_ + permK(q * 4)) = pk;   // quad stays contiguous
    }

    // ---- embedding GEMM: h0 = X @ embW + embB (A in regs, in-place write)
    {
      const bf16_t* rowp = myh + c_lane * H_;
      v16bf a0 = load_fragA(rowp, 0, lane);
      v16bf a1 = load_fragA(rowp, 1, lane);
      #pragma unroll
      for (int nt = 0; nt < 8; ++nt) {
        v8f acc = {};
        acc = __builtin_amdgcn_wmma_f32_16x16x32_bf16(false, a0, false,
                load_fragB(embT, F_, nt, 0, lane), (short)0, acc, false, false);
        acc = __builtin_amdgcn_wmma_f32_16x16x32_bf16(false, a1, false,
                load_fragB(embT, F_, nt, 1, lane), (short)0, acc, false, false);
        int   col  = nt * 16 + c_lane;
        int   pcol = permK(col);
        float bias = biasL[col];
        #pragma unroll
        for (int r = 0; r < 8; ++r)
          store_bf16(myh + (rbase + r) * H_ + pcol, acc[r] + bias);
      }
    }

    // ---- gate layers (weights + biases resident in LDS, no barriers) ----
    for (int l = 0; l < L_; ++l) {
      const bf16_t* gWiT = wlds + H_ * F_ + (l * 3 + 0) * (H_ * H_);
      const bf16_t* gWoT = wlds + H_ * F_ + (l * 3 + 1) * (H_ * H_);
      const bf16_t* gWcT = wlds + H_ * F_ + (l * 3 + 2) * (H_ * H_);
      const float*  bI   = biasL + H_ + (l * 3 + 0) * H_;
      const float*  bO   = biasL + H_ + (l * 3 + 1) * H_;
      const float*  bC   = biasL + H_ + (l * 3 + 2) * H_;

      // full-K A fragments into registers, then write h' in place
      const bf16_t* rowp = myh + c_lane * H_;
      v16bf A[4];
      #pragma unroll
      for (int ks = 0; ks < 4; ++ks) A[ks] = load_fragA(rowp, ks, lane);

      for (int nt = 0; nt < 8; ++nt) {
        v8f ai = {}, ao = {}, ac = {};
        #pragma unroll
        for (int ks = 0; ks < 4; ++ks) {
          v16bf bi = load_fragB(gWiT, H_, nt, ks, lane);
          v16bf bo = load_fragB(gWoT, H_, nt, ks, lane);
          v16bf bc = load_fragB(gWcT, H_, nt, ks, lane);
          ai = __builtin_amdgcn_wmma_f32_16x16x32_bf16(false, A[ks], false, bi, (short)0, ai, false, false);
          ao = __builtin_amdgcn_wmma_f32_16x16x32_bf16(false, A[ks], false, bo, (short)0, ao, false, false);
          ac = __builtin_amdgcn_wmma_f32_16x16x32_bf16(false, A[ks], false, bc, (short)0, ac, false, false);
        }
        int   col  = nt * 16 + c_lane;
        float bi_s = bI[col];
        float bo_s = bO[col];
        float bc_s = bC[col];
        if (l == L_ - 1) {
          float csum = 0.0f;
          #pragma unroll
          for (int r = 0; r < 8; ++r) {
            float iv = sigm(ai[r] + bi_s);
            float ov = sigm(ao[r] + bo_s);
            float cv = tanh_(ac[r] + bc_s);
            csum += ov * tanh_(iv * cv);
          }
          wsum[nt] += csum;                  // in-register over rows & iters
        } else {
          int pcol = permK(col);
          #pragma unroll
          for (int r = 0; r < 8; ++r) {
            float iv = sigm(ai[r] + bi_s);
            float ov = sigm(ao[r] + bo_s);
            float cv = tanh_(ac[r] + bc_s);
            store_bf16(myh + (rbase + r) * H_ + pcol, ov * tanh_(iv * cv));
          }
        }
      }
    }
  }

  // ---- deterministic batch-mean reduction (fixed summation order) ----
  #pragma unroll
  for (int nt = 0; nt < 8; ++nt)
    pacc[(wave * 2 + (lane >= 16 ? 1 : 0)) * H_ + nt * 16 + c_lane] = wsum[nt];
  __syncthreads();
  if (tid < H_) {
    float s = 0.0f;
    #pragma unroll
    for (int j = 0; j < WAVES * 2; ++j) s += pacc[j * H_ + tid];
    out[(size_t)b * H_ + tid] = s * (1.0f / (float)N_);
  }
}

extern "C" void kernel_launch(void* const* d_in, const int* in_sizes, int n_in,
                              void* d_out, int out_size, void* d_ws, size_t ws_size,
                              hipStream_t stream) {
  (void)in_sizes; (void)n_in; (void)ws_size; (void)out_size;
  const float* feat = (const float*)d_in[0];
  const float* embW = (const float*)d_in[1];
  const float* embB = (const float*)d_in[2];
  const float* Wi   = (const float*)d_in[3];
  const float* Wo   = (const float*)d_in[4];
  const float* Wc   = (const float*)d_in[5];
  const float* bWi  = (const float*)d_in[6];
  const float* bUi  = (const float*)d_in[7];
  const float* bWo  = (const float*)d_in[8];
  const float* bUo  = (const float*)d_in[9];
  const float* bWc  = (const float*)d_in[10];
  const float* bUc  = (const float*)d_in[11];
  float* out = (float*)d_out;
  bf16_t* wpack = (bf16_t*)d_ws;            // 216.6 KB used

  int prep_elems = WELEMS + NBIAS;
  prep_weights<<<(prep_elems + 255) / 256, 256, 0, stream>>>(
      embW, embB, Wi, Wo, Wc, bWi, bUi, bWo, bUo, bWc, bUc, wpack);
  tree_encoder_fused<<<B_, THREADS, 0, stream>>>(feat, wpack, out);
}
